// CaPa_MoE_clinical_MLP_31379031065169
// MI455X (gfx1250) — matrix operand order repaired and checked by reading, compiled, not run on hardware
//
#include <hip/hip_runtime.h>
#include <hip/hip_bf16.h>
#include <math.h>

typedef __bf16 bf16_t;
typedef __attribute__((ext_vector_type(16))) __bf16 v16bf;
typedef __attribute__((ext_vector_type(8)))  __bf16 v8bf;
typedef __attribute__((ext_vector_type(8)))  float  v8f;
typedef __attribute__((ext_vector_type(4)))  unsigned int u32x4;
typedef __attribute__((ext_vector_type(8)))  int i32x8;
typedef __attribute__((ext_vector_type(4)))  int i32x4;

namespace {

constexpr int kN1 = 30000, kN2 = 30000;
constexpr int kVIR = 2560, kEMB = 1024, kH = 512, kD = 256, kNC = 2;
constexpr int kChunks = 120;

// LDS carve (bytes) for fused CLAM kernels: 64-row M tile, 4 M-groups x 8 N-waves.
// A tiles are f32, TDM-loaded, double buffered, padded rows of 36 dwords
// (bank-conflict free, 16B aligned).
constexpr int AT_LD = 36;                        // f32 elements per padded row
constexpr int AT_TILE_BYTES = 64 * AT_LD * 4;    // 9216
constexpr int AF_BYTES = 2 * AT_TILE_BYTES;      // 18432 (double buffer)
constexpr int XS_LD = kEMB + 8;                  // 1032 (stage-1 output, bf16)
constexpr int HS_LD = kH + 8;                    // 520  (hp tile, bf16)
constexpr int GS_LD = kD + 8;                    // 264  (gated tile, bf16)
constexpr int XS_BYTES = 64 * XS_LD * 2;         // 132096
constexpr int HS_BYTES = 64 * HS_LD * 2;         // 66560
constexpr int GS_BYTES = 64 * GS_LD * 2;         // 33792
constexpr int SMEM_V = AF_BYTES + XS_BYTES + HS_BYTES + GS_BYTES;  // 250880 (<320KB)
constexpr int SMEM_U = AF_BYTES + HS_BYTES + GS_BYTES;             // 118784

// ------------------------- TDM: async 2D f32 tile -> LDS --------------------
// Loads a [rows<=64] x [32 f32] tile starting at gptr (row stride row_stride
// elements) into LDS at byte offset lds_off, padding each 32-dword row by 4
// dwords (LDS row stride = 36 dwords). Rows beyond rows_left are zero-filled.
// Issued once per wave; tracked with TENSORcnt.

__device__ __forceinline__ void tdm_load_a_tile(const float* gptr,
                                                unsigned lds_off,
                                                unsigned rows_left,
                                                unsigned row_stride) {
  unsigned long long ga = (unsigned long long)(const void*)gptr;
  u32x4 g0;
  g0[0] = 1u;                                            // count=1 valid user D#
  g0[1] = lds_off;                                       // lds_addr (bytes)
  g0[2] = (unsigned)(ga & 0xFFFFFFFFu);                  // global_addr[31:0]
  g0[3] = (unsigned)((ga >> 32) & 0x01FFFFFFu) | (2u << 30);  // addr[56:32], type=2
  const unsigned td0 = 32u;            // tensor_dim0 (tile always in-range in x)
  const unsigned td1 = rows_left;      // tensor_dim1: zero-fill rows beyond this
  i32x8 g1;
  g1[0] = (int)((2u << 16)     // data_size = 4B
                | (1u << 20)   // pad_enable
                | (4u << 22)   // pad_interval code 4 -> every 32 dwords
                | (3u << 25)); // pad_amount  code 3 -> 4 dwords
  g1[1] = (int)((td0 & 0xFFFFu) << 16);                       // tensor_dim0[15:0]
  g1[2] = (int)((td0 >> 16) | ((td1 & 0xFFFFu) << 16));       // dim0 hi | dim1 lo
  g1[3] = (int)((td1 >> 16) | (32u << 16));                   // dim1 hi | tile_dim0
  g1[4] = (int)(64u);                                         // tile_dim1 | tile_dim2=0
  g1[5] = (int)row_stride;                                    // tensor_dim0_stride lo
  g1[6] = 0;                                                  // stride0 hi | stride1 lo
  g1[7] = 0;
  i32x4 z = {0, 0, 0, 0};
  i32x8 z8 = {0, 0, 0, 0, 0, 0, 0, 0};
  __builtin_amdgcn_tensor_load_to_lds(g0, g1, z, z, z8, 0);
}

// ------------------------- WMMA helpers -------------------------------------

__device__ __forceinline__ v8f wmma_bf16(const v16bf& a, const v16bf& b, const v8f& c) {
  // D(f32 16x16) = A(16x32 bf16) * B(32x16 bf16) + C
  return __builtin_amdgcn_wmma_f32_16x16x32_bf16(false, a, false, b, (short)0, c,
                                                 false, false);
}

// A fragment from an LDS row-major bf16 tile (ld in elements).
// Lane<16: M=lane, K=k0+{0..7,16..23}; lane>=16: M=lane-16, K=k0+{8..15,24..31}
__device__ __forceinline__ v16bf frag_a_lds(const bf16_t* lds, int ld, int k0) {
  const int lane = threadIdx.x & 31;
  const int m = lane & 15;
  const int kh = (lane & 16) ? 8 : 0;
  const bf16_t* p = lds + m * ld + k0 + kh;
  v8bf lo = *reinterpret_cast<const v8bf*>(p);
  v8bf hi = *reinterpret_cast<const v8bf*>(p + 16);
  v16bf a;
#pragma unroll
  for (int i = 0; i < 8; ++i) { a[i] = lo[i]; a[8 + i] = hi[i]; }
  return a;
}

// Same fragment but from an f32 LDS tile (TDM-staged input), converting to bf16.
__device__ __forceinline__ v16bf frag_a_lds_f32(const float* lds, int ld) {
  const int lane = threadIdx.x & 31;
  const int m = lane & 15;
  const int kh = (lane & 16) ? 8 : 0;
  const float* p = lds + m * ld + kh;
  v16bf a;
#pragma unroll
  for (int i = 0; i < 8; ++i) { a[i] = (bf16_t)p[i]; a[8 + i] = (bf16_t)p[16 + i]; }
  return a;
}

// B fragment from pre-transposed bf16 weights Wt[N][K] (row-major, 32B rows).
// Lane<16: N=n0+lane, K=k0+{0..7,16..23} ; lane>=16: N=n0+lane-16, K=k0+{8..15,24..31}
__device__ __forceinline__ v16bf frag_b_gl(const bf16_t* __restrict__ Wt, int K,
                                           int n0, int k0) {
  const int lane = threadIdx.x & 31;
  const int n = n0 + (lane & 15);
  const int kk = k0 + ((lane & 16) ? 8 : 0);
  const bf16_t* p = Wt + (size_t)n * K + kk;
  v8bf lo = *reinterpret_cast<const v8bf*>(p);
  v8bf hi = *reinterpret_cast<const v8bf*>(p + 16);
  v16bf b;
#pragma unroll
  for (int i = 0; i < 8; ++i) { b[i] = lo[i]; b[8 + i] = hi[i]; }
  return b;
}

// ------------------------- weight convert+transpose -------------------------

__global__ __launch_bounds__(256) void k_convT(const float* __restrict__ W,
                                               bf16_t* __restrict__ Wt,
                                               int K, int Nn) {
  // W: [K, Nn] f32 row-major  ->  Wt: [Nn, K] bf16 row-major
  size_t i = (size_t)blockIdx.x * 256 + threadIdx.x;
  if (i < (size_t)K * Nn) {
    int k = (int)(i / Nn), n = (int)(i % Nn);
    Wt[(size_t)n * K + k] = (bf16_t)W[i];
  }
}

// ------------------------- fused CLAM (virchow path) ------------------------
// per 64-row tile: x = X@Wv+bv -> hp = relu(x@Wfc+bfc) -> g = tanh(.)*sigm(.)
// -> A_raw = g@Wc+bc.  hp spilled bf16 for the pooling pass.

__global__ __launch_bounds__(1024) void k_clam_v(
    const float* __restrict__ X, const bf16_t* __restrict__ WvT,
    const float* __restrict__ bv, const bf16_t* __restrict__ WfcT,
    const float* __restrict__ bfc, const bf16_t* __restrict__ WaT,
    const float* __restrict__ ba, const bf16_t* __restrict__ WbT,
    const float* __restrict__ bb, const float* __restrict__ Wc,
    const float* __restrict__ bc, bf16_t* __restrict__ hp_out,
    float* __restrict__ A_out, int N) {
  extern __shared__ char smem[];
  float*  af32 = (float*)smem;                             // 2 x [64][36] f32
  bf16_t* xs = (bf16_t*)(smem + AF_BYTES);                 // [64][1032]
  bf16_t* hs = (bf16_t*)(smem + AF_BYTES + XS_BYTES);      // [64][520]
  bf16_t* gs = (bf16_t*)(smem + AF_BYTES + XS_BYTES + HS_BYTES);  // [64][264]

  const int row0 = blockIdx.x * 64;
  const int wave = threadIdx.x >> 5;
  const int lane = threadIdx.x & 31;
  const int mg = wave >> 3;         // M-group (0..3): rows mg*16..mg*16+15
  const int ng = wave & 7;          // N-wave  (0..7)
  const int mrow = (lane >> 4) << 3;
  const int ncol = lane & 15;
  const bool issuer = (threadIdx.x < 32);
  const unsigned rows_left = (unsigned)(N - row0);

  // ---- stage 1: x[64,1024] = X[64,2560] @ Wv^T, bf16 WMMA, f32 acc ----------
  if (issuer)
    tdm_load_a_tile(X + (size_t)row0 * kVIR, 0u, rows_left, kVIR);
  v8f acc1[8] = {};
  for (int k0 = 0, it = 0; k0 < kVIR; k0 += 32, ++it) {
    if (issuer) __builtin_amdgcn_s_wait_tensorcnt(0);
    __syncthreads();  // tile (it&1) ready; previous readers of other buf done
    if (issuer && (k0 + 32 < kVIR))
      tdm_load_a_tile(X + (size_t)row0 * kVIR + (k0 + 32),
                      (unsigned)(((it + 1) & 1) * AT_TILE_BYTES), rows_left, kVIR);
    const float* atile = af32 + (it & 1) * (64 * AT_LD) + mg * 16 * AT_LD;
    v16bf af = frag_a_lds_f32(atile, AT_LD);
    v16bf bcur = frag_b_gl(WvT, kVIR, ng * 128, k0);
#pragma unroll
    for (int t = 0; t < 8; ++t) {
      v16bf bnxt = bcur;
      if (t < 7) bnxt = frag_b_gl(WvT, kVIR, ng * 128 + (t + 1) * 16, k0);
      acc1[t] = wmma_bf16(af, bcur, acc1[t]);
      bcur = bnxt;
    }
  }
  __syncthreads();
#pragma unroll
  for (int t = 0; t < 8; ++t) {
    int n = ng * 128 + t * 16 + ncol;
    float bias = bv[n];
#pragma unroll
    for (int i = 0; i < 8; ++i)
      xs[(mg * 16 + mrow + i) * XS_LD + n] = (bf16_t)(acc1[t][i] + bias);
  }
  __syncthreads();

  // ---- stage 2: hp[64,512] = relu(x @ Wfc^T) --------------------------------
  v8f acc2[4] = {};
  const bf16_t* xmg = xs + mg * 16 * XS_LD;
  for (int k0 = 0; k0 < kEMB; k0 += 32) {
    v16bf af = frag_a_lds(xmg, XS_LD, k0);
    v16bf bcur = frag_b_gl(WfcT, kEMB, ng * 64, k0);
#pragma unroll
    for (int t = 0; t < 4; ++t) {
      v16bf bnxt = bcur;
      if (t < 3) bnxt = frag_b_gl(WfcT, kEMB, ng * 64 + (t + 1) * 16, k0);
      acc2[t] = wmma_bf16(af, bcur, acc2[t]);
      bcur = bnxt;
    }
  }
#pragma unroll
  for (int t = 0; t < 4; ++t) {
    int n = ng * 64 + t * 16 + ncol;
    float bias = bfc[n];
#pragma unroll
    for (int i = 0; i < 8; ++i) {
      int grow = row0 + mg * 16 + mrow + i;
      bf16_t hv = (bf16_t)fmaxf(acc2[t][i] + bias, 0.f);
      hs[(mg * 16 + mrow + i) * HS_LD + n] = hv;
      if (grow < N) hp_out[(size_t)grow * kH + n] = hv;
    }
  }
  __syncthreads();

  // ---- stage 3: g = tanh(hp@Wa^T+ba) * sigmoid(hp@Wb^T+bb) ------------------
  v8f accA[2] = {}, accB[2] = {};
  const bf16_t* hmg = hs + mg * 16 * HS_LD;
  for (int k0 = 0; k0 < kH; k0 += 32) {
    v16bf af = frag_a_lds(hmg, HS_LD, k0);
#pragma unroll
    for (int t = 0; t < 2; ++t) {
      accA[t] = wmma_bf16(af, frag_b_gl(WaT, kH, ng * 32 + t * 16, k0), accA[t]);
      accB[t] = wmma_bf16(af, frag_b_gl(WbT, kH, ng * 32 + t * 16, k0), accB[t]);
    }
  }
#pragma unroll
  for (int t = 0; t < 2; ++t) {
    int n = ng * 32 + t * 16 + ncol;
    float biasA = ba[n], biasB = bb[n];
#pragma unroll
    for (int i = 0; i < 8; ++i) {
      float av = tanhf(accA[t][i] + biasA);
      float sv = 1.f / (1.f + __expf(-(accB[t][i] + biasB)));
      gs[(mg * 16 + mrow + i) * GS_LD + n] = (bf16_t)(av * sv);
    }
  }
  __syncthreads();

  // ---- stage 4: A_raw[64,2] = g @ Wc + bc  ->  A_out is [NC][N] -------------
  if (threadIdx.x < 128) {
    int r = threadIdx.x & 63, c = threadIdx.x >> 6;
    if (row0 + r < N) {
      float s = bc[c];
      for (int k = 0; k < kD; ++k) s += (float)gs[r * GS_LD + k] * Wc[k * kNC + c];
      A_out[(size_t)c * N + row0 + r] = s;
    }
  }
}

// ------------------------- fused CLAM (UNI path, no projection) -------------

__global__ __launch_bounds__(1024) void k_clam_u(
    const float* __restrict__ Y, const bf16_t* __restrict__ WfcT,
    const float* __restrict__ bfc, const bf16_t* __restrict__ WaT,
    const float* __restrict__ ba, const bf16_t* __restrict__ WbT,
    const float* __restrict__ bb, const float* __restrict__ Wc,
    const float* __restrict__ bc, bf16_t* __restrict__ hp_out,
    float* __restrict__ A_out, int N) {
  extern __shared__ char smem[];
  float*  af32 = (float*)smem;
  bf16_t* hs = (bf16_t*)(smem + AF_BYTES);
  bf16_t* gs = (bf16_t*)(smem + AF_BYTES + HS_BYTES);

  const int row0 = blockIdx.x * 64;
  const int wave = threadIdx.x >> 5;
  const int lane = threadIdx.x & 31;
  const int mg = wave >> 3, ng = wave & 7;
  const int mrow = (lane >> 4) << 3, ncol = lane & 15;
  const bool issuer = (threadIdx.x < 32);
  const unsigned rows_left = (unsigned)(N - row0);

  // hp = relu(Y @ Wfc^T)
  if (issuer)
    tdm_load_a_tile(Y + (size_t)row0 * kEMB, 0u, rows_left, kEMB);
  v8f acc2[4] = {};
  for (int k0 = 0, it = 0; k0 < kEMB; k0 += 32, ++it) {
    if (issuer) __builtin_amdgcn_s_wait_tensorcnt(0);
    __syncthreads();
    if (issuer && (k0 + 32 < kEMB))
      tdm_load_a_tile(Y + (size_t)row0 * kEMB + (k0 + 32),
                      (unsigned)(((it + 1) & 1) * AT_TILE_BYTES), rows_left, kEMB);
    const float* atile = af32 + (it & 1) * (64 * AT_LD) + mg * 16 * AT_LD;
    v16bf af = frag_a_lds_f32(atile, AT_LD);
    v16bf bcur = frag_b_gl(WfcT, kEMB, ng * 64, k0);
#pragma unroll
    for (int t = 0; t < 4; ++t) {
      v16bf bnxt = bcur;
      if (t < 3) bnxt = frag_b_gl(WfcT, kEMB, ng * 64 + (t + 1) * 16, k0);
      acc2[t] = wmma_bf16(af, bcur, acc2[t]);
      bcur = bnxt;
    }
  }
  __syncthreads();
#pragma unroll
  for (int t = 0; t < 4; ++t) {
    int n = ng * 64 + t * 16 + ncol;
    float bias = bfc[n];
#pragma unroll
    for (int i = 0; i < 8; ++i) {
      int grow = row0 + mg * 16 + mrow + i;
      bf16_t hv = (bf16_t)fmaxf(acc2[t][i] + bias, 0.f);
      hs[(mg * 16 + mrow + i) * HS_LD + n] = hv;
      if (grow < N) hp_out[(size_t)grow * kH + n] = hv;
    }
  }
  __syncthreads();

  v8f accA[2] = {}, accB[2] = {};
  const bf16_t* hmg = hs + mg * 16 * HS_LD;
  for (int k0 = 0; k0 < kH; k0 += 32) {
    v16bf af = frag_a_lds(hmg, HS_LD, k0);
#pragma unroll
    for (int t = 0; t < 2; ++t) {
      accA[t] = wmma_bf16(af, frag_b_gl(WaT, kH, ng * 32 + t * 16, k0), accA[t]);
      accB[t] = wmma_bf16(af, frag_b_gl(WbT, kH, ng * 32 + t * 16, k0), accB[t]);
    }
  }
#pragma unroll
  for (int t = 0; t < 2; ++t) {
    int n = ng * 32 + t * 16 + ncol;
    float biasA = ba[n], biasB = bb[n];
#pragma unroll
    for (int i = 0; i < 8; ++i) {
      float av = tanhf(accA[t][i] + biasA);
      float sv = 1.f / (1.f + __expf(-(accB[t][i] + biasB)));
      gs[(mg * 16 + mrow + i) * GS_LD + n] = (bf16_t)(av * sv);
    }
  }
  __syncthreads();

  if (threadIdx.x < 128) {
    int r = threadIdx.x & 63, c = threadIdx.x >> 6;
    if (row0 + r < N) {
      float s = bc[c];
      for (int k = 0; k < kD; ++k) s += (float)gs[r * GS_LD + k] * Wc[k * kNC + c];
      A_out[(size_t)c * N + row0 + r] = s;
    }
  }
}

// ------------------------- softmax pooling (deterministic) ------------------

__global__ __launch_bounds__(256) void k_colmax(const float* __restrict__ A, int N,
                                                float* __restrict__ mx) {
  __shared__ float red[256];
  int c = blockIdx.x;
  float m = -3.4e38f;
  for (int i = threadIdx.x; i < N; i += 256) m = fmaxf(m, A[(size_t)c * N + i]);
  red[threadIdx.x] = m;
  __syncthreads();
  for (int s = 128; s > 0; s >>= 1) {
    if (threadIdx.x < s) red[threadIdx.x] = fmaxf(red[threadIdx.x], red[threadIdx.x + s]);
    __syncthreads();
  }
  if (threadIdx.x == 0) mx[c] = red[0];
}

// Per-chunk partials of Z_c = sum exp(A-max) and Macc[c][h] = sum exp * hp[n][h]
__global__ __launch_bounds__(256) void k_attn_accum(
    const float* __restrict__ A, const float* __restrict__ mx,
    const bf16_t* __restrict__ hp, int N, int rpc, float* __restrict__ part) {
  int n0 = blockIdx.x * rpc;
  int n1 = n0 + rpc; if (n1 > N) n1 = N;
  const float m0 = mx[0], m1 = mx[1];
  const int h0 = threadIdx.x, h1 = threadIdx.x + 256;
  float a00 = 0, a01 = 0, a10 = 0, a11 = 0, z0 = 0, z1 = 0;
  for (int n = n0; n < n1; ++n) {
    float w0 = __expf(A[n] - m0);
    float w1 = __expf(A[(size_t)N + n] - m1);
    float v0 = (float)hp[(size_t)n * kH + h0];
    float v1 = (float)hp[(size_t)n * kH + h1];
    a00 += w0 * v0; a01 += w0 * v1;
    a10 += w1 * v0; a11 += w1 * v1;
    z0 += w0; z1 += w1;
  }
  float* p = part + (size_t)blockIdx.x * (kNC * kH + kNC);
  p[h0] = a00; p[h1] = a01;
  p[kH + h0] = a10; p[kH + h1] = a11;
  if (threadIdx.x == 0) { p[kNC * kH] = z0; p[kNC * kH + 1] = z1; }
}

__global__ __launch_bounds__(256) void k_reduce_part(const float* __restrict__ part,
                                                     float* __restrict__ Macc,
                                                     float* __restrict__ Z) {
  const int STRIDE = kNC * kH + kNC;  // 1026
  int i = blockIdx.x * 256 + threadIdx.x;
  if (i < kNC * kH) {
    float s = 0;
    for (int ch = 0; ch < kChunks; ++ch) s += part[(size_t)ch * STRIDE + i];
    Macc[i] = s;
  } else if (i < STRIDE) {
    float s = 0;
    for (int ch = 0; ch < kChunks; ++ch) s += part[(size_t)ch * STRIDE + i];
    Z[i - kNC * kH] = s;
  }
}

// ------------------------- head: gate / MoE / fusion / classifier ----------

__global__ __launch_bounds__(256) void k_head(
    const float* __restrict__ Maccv, const float* __restrict__ Zv,
    const float* __restrict__ Maccu, const float* __restrict__ Zu,
    const float* __restrict__ clinical, const float* __restrict__ clinb,
    const float* __restrict__ clinw, const float* __restrict__ g1b,
    const float* __restrict__ g1w, const float* __restrict__ g2b,
    const float* __restrict__ g2w, const float* __restrict__ e1l1b,
    const float* __restrict__ e1l1w, const float* __restrict__ e1l2b,
    const float* __restrict__ e1l2w, const float* __restrict__ e2l1b,
    const float* __restrict__ e2l1w, const float* __restrict__ e2l2b,
    const float* __restrict__ e2l2w, const float* __restrict__ e2ob,
    const float* __restrict__ e2ow, const float* __restrict__ e3l1b,
    const float* __restrict__ e3l1w, const float* __restrict__ e3l2b,
    const float* __restrict__ e3l2w, const float* __restrict__ fub,
    const float* __restrict__ fuw, const float* __restrict__ clsb,
    const float* __restrict__ clsw, float* __restrict__ out) {
  __shared__ float cvec[kH];
  __shared__ float Mv[kNC][kH], Mu[kNC][kH];
  __shared__ float gin[3 * kH];
  __shared__ float gd[kD];
  __shared__ float wsel[4];
  __shared__ float t1[kNC][4 * kH];
  __shared__ float x2[kNC][2 * kH];
  __shared__ float moe[kNC][kH];
  __shared__ float fus[kNC][kH];
  const int tid = threadIdx.x;

  for (int h = tid; h < kH; h += 256) {
    float s = clinb[h];
#pragma unroll
    for (int k = 0; k < 6; ++k) s += clinical[k] * clinw[k * kH + h];
    cvec[h] = fmaxf(s, 0.f);
    for (int c = 0; c < kNC; ++c) {
      Mv[c][h] = Maccv[c * kH + h] / Zv[c];
      Mu[c][h] = Maccu[c * kH + h] / Zu[c];
    }
  }
  __syncthreads();
  for (int h = tid; h < kH; h += 256) {
    gin[h] = 0.5f * (Mv[0][h] + Mv[1][h]);
    gin[kH + h] = 0.5f * (Mu[0][h] + Mu[1][h]);
    gin[2 * kH + h] = cvec[h];
  }
  __syncthreads();
  for (int j = tid; j < kD; j += 256) {
    float s = g1b[j];
    for (int k = 0; k < 3 * kH; ++k) s += gin[k] * g1w[k * kD + j];
    gd[j] = fmaxf(s, 0.f);
  }
  __syncthreads();
  if (tid == 0) {
    float lg[3];
    for (int j = 0; j < 3; ++j) {
      float s = g2b[j];
      for (int k = 0; k < kD; ++k) s += gd[k] * g2w[k * 3 + j];
      // deterministic pseudo-Gumbel noise (stand-in for jax.random.key(42))
      unsigned x = 42u * 0x9E3779B9u + (unsigned)j * 0x85EBCA6Bu;
      x ^= x >> 16; x *= 0x7FEB352Du; x ^= x >> 15; x *= 0x846CA68Bu; x ^= x >> 16;
      float U = (float)(x >> 8) * (1.0f / 16777216.0f);
      float g = -logf(-logf(U + 1e-20f) + 1e-20f);
      lg[j] = s + g;  // tau = 1
    }
    int idx = 0;
    if (lg[1] > lg[idx]) idx = 1;
    if (lg[2] > lg[idx]) idx = 2;
    // straight-through forward value is the one-hot
    wsel[0] = (idx == 0) ? 1.f : 0.f;
    wsel[1] = (idx == 1) ? 1.f : 0.f;
    wsel[2] = (idx == 2) ? 1.f : 0.f;
  }
  __syncthreads();

  // E1 (Mv) and E3 (Mu): H -> 2H -> H, relu/relu
  for (int c = 0; c < kNC; ++c)
    for (int j = tid; j < 2 * kH; j += 256) {
      float s1 = e1l1b[j], s3 = e3l1b[j];
      for (int k = 0; k < kH; ++k) {
        s1 += Mv[c][k] * e1l1w[k * 2 * kH + j];
        s3 += Mu[c][k] * e3l1w[k * 2 * kH + j];
      }
      t1[c][j] = fmaxf(s1, 0.f);
      t1[c][2 * kH + j] = fmaxf(s3, 0.f);
    }
  __syncthreads();
  for (int c = 0; c < kNC; ++c)
    for (int j = tid; j < kH; j += 256) {
      float s1 = e1l2b[j], s3 = e3l2b[j];
      for (int k = 0; k < 2 * kH; ++k) {
        s1 += t1[c][k] * e1l2w[k * kH + j];
        s3 += t1[c][2 * kH + k] * e3l2w[k * kH + j];
      }
      moe[c][j] = wsel[0] * fmaxf(s1, 0.f) + wsel[2] * fmaxf(s3, 0.f);
      x2[c][j] = Mv[c][j];
      x2[c][kH + j] = Mu[c][j];
    }
  __syncthreads();
  // E2: 2H -> 4H -> 2H -> (e2_out) H
  for (int c = 0; c < kNC; ++c)
    for (int j = tid; j < 4 * kH; j += 256) {
      float s = e2l1b[j];
      for (int k = 0; k < 2 * kH; ++k) s += x2[c][k] * e2l1w[k * 4 * kH + j];
      t1[c][j] = fmaxf(s, 0.f);
    }
  __syncthreads();
  for (int c = 0; c < kNC; ++c)
    for (int j = tid; j < 2 * kH; j += 256) {
      float s = e2l2b[j];
      for (int k = 0; k < 4 * kH; ++k) s += t1[c][k] * e2l2w[k * 2 * kH + j];
      x2[c][j] = fmaxf(s, 0.f);
    }
  __syncthreads();
  for (int c = 0; c < kNC; ++c)
    for (int j = tid; j < kH; j += 256) {
      float s = e2ob[j];
      for (int k = 0; k < 2 * kH; ++k) s += x2[c][k] * e2ow[k * kH + j];
      moe[c][j] += wsel[1] * s;
    }
  __syncthreads();
  // fusion([moe, cvec]) -> H
  for (int c = 0; c < kNC; ++c)
    for (int j = tid; j < kH; j += 256) {
      float s = fub[j];
      for (int k = 0; k < kH; ++k) s += moe[c][k] * fuw[k * kH + j];
      for (int k = 0; k < kH; ++k) s += cvec[k] * fuw[(kH + k) * kH + j];
      fus[c][j] = s;
    }
  __syncthreads();
  if (tid < kNC) {
    float s = clsb[tid];
    for (int h = 0; h < kH; ++h) s += fus[tid][h] * clsw[tid * kH + h];
    out[tid] = s;
  }
  __syncthreads();
  if (tid == 0) {
    float l0 = out[0], l1 = out[1];
    float mx = fmaxf(l0, l1);
    float e0 = expf(l0 - mx), e1 = expf(l1 - mx);
    out[2] = e0 / (e0 + e1);
    out[3] = e1 / (e0 + e1);
    out[4] = (l1 > l0) ? 1.0f : 0.0f;  // Y_hat
  }
}

}  // namespace

// ----------------------------------------------------------------------------

extern "C" void kernel_launch(void* const* d_in, const int* in_sizes, int n_in,
                              void* d_out, int out_size, void* d_ws, size_t ws_size,
                              hipStream_t stream) {
  (void)in_sizes; (void)n_in; (void)out_size; (void)ws_size;

  // inputs: h_virchow, h_UNI, clinical, then params leaves in jax pytree
  // (recursively key-sorted) order.
  const float* X        = (const float*)d_in[0];   // [N1, VIR]
  const float* Y        = (const float*)d_in[1];   // [N2, EMB]
  const float* clinical = (const float*)d_in[2];   // [6]
  const float* au_a_b  = (const float*)d_in[3];
  const float* au_a_w  = (const float*)d_in[4];
  const float* au_b_b  = (const float*)d_in[5];
  const float* au_b_w  = (const float*)d_in[6];
  const float* au_c_b  = (const float*)d_in[7];
  const float* au_c_w  = (const float*)d_in[8];
  const float* au_fc_b = (const float*)d_in[9];
  const float* au_fc_w = (const float*)d_in[10];
  const float* av_a_b  = (const float*)d_in[11];
  const float* av_a_w  = (const float*)d_in[12];
  const float* av_b_b  = (const float*)d_in[13];
  const float* av_b_w  = (const float*)d_in[14];
  const float* av_c_b  = (const float*)d_in[15];
  const float* av_c_w  = (const float*)d_in[16];
  const float* av_fc_b = (const float*)d_in[17];
  const float* av_fc_w = (const float*)d_in[18];
  const float* clin_b  = (const float*)d_in[19];
  const float* clin_w  = (const float*)d_in[20];
  const float* cls_b   = (const float*)d_in[21];
  const float* cls_w   = (const float*)d_in[22];
  const float* e1l1_b  = (const float*)d_in[23];
  const float* e1l1_w  = (const float*)d_in[24];
  const float* e1l2_b  = (const float*)d_in[25];
  const float* e1l2_w  = (const float*)d_in[26];
  const float* e2l1_b  = (const float*)d_in[27];
  const float* e2l1_w  = (const float*)d_in[28];
  const float* e2l2_b  = (const float*)d_in[29];
  const float* e2l2_w  = (const float*)d_in[30];
  const float* e2o_b   = (const float*)d_in[31];
  const float* e2o_w   = (const float*)d_in[32];
  const float* e3l1_b  = (const float*)d_in[33];
  const float* e3l1_w  = (const float*)d_in[34];
  const float* e3l2_b  = (const float*)d_in[35];
  const float* e3l2_w  = (const float*)d_in[36];
  const float* fu_b    = (const float*)d_in[37];
  const float* fu_w    = (const float*)d_in[38];
  const float* g1_b    = (const float*)d_in[39];
  const float* g1_w    = (const float*)d_in[40];
  const float* g2_b    = (const float*)d_in[41];
  const float* g2_w    = (const float*)d_in[42];
  const float* pv_b    = (const float*)d_in[43];
  const float* pv_w    = (const float*)d_in[44];

  // d_out: [logits(2) | Y_prob(2) | Y_hat(1) | A_v(2*N1) | A_u(2*N2)]
  float* out = (float*)d_out;
  float* Av = out + 5;
  float* Au = out + 5 + kNC * kN1;

  // workspace carve
  size_t off = 0;
  auto carve = [&](size_t bytes) -> void* {
    void* p = (char*)d_ws + off;
    off += (bytes + 255) & ~(size_t)255;
    return p;
  };
  bf16_t* WvT   = (bf16_t*)carve((size_t)kEMB * kVIR * 2);
  bf16_t* WfcvT = (bf16_t*)carve((size_t)kH * kEMB * 2);
  bf16_t* WfcuT = (bf16_t*)carve((size_t)kH * kEMB * 2);
  bf16_t* WavT  = (bf16_t*)carve((size_t)kD * kH * 2);
  bf16_t* WbvT  = (bf16_t*)carve((size_t)kD * kH * 2);
  bf16_t* WauT  = (bf16_t*)carve((size_t)kD * kH * 2);
  bf16_t* WbuT  = (bf16_t*)carve((size_t)kD * kH * 2);
  bf16_t* hpv   = (bf16_t*)carve((size_t)kN1 * kH * 2);
  bf16_t* hpu   = (bf16_t*)carve((size_t)kN2 * kH * 2);
  float* mxv    = (float*)carve(kNC * 4);
  float* mxu    = (float*)carve(kNC * 4);
  float* Maccv  = (float*)carve(kNC * kH * 4);
  float* Zv     = (float*)carve(kNC * 4);
  float* Maccu  = (float*)carve(kNC * kH * 4);
  float* Zu     = (float*)carve(kNC * 4);
  float* partv  = (float*)carve((size_t)kChunks * (kNC * kH + kNC) * 4);
  float* partu  = (float*)carve((size_t)kChunks * (kNC * kH + kNC) * 4);

  auto cdiv = [](long a, long b) { return (int)((a + b - 1) / b); };

  // 1) weight convert + transpose to bf16 [N][K]
  k_convT<<<cdiv((long)kVIR * kEMB, 256), 256, 0, stream>>>(pv_w, WvT, kVIR, kEMB);
  k_convT<<<cdiv((long)kEMB * kH, 256), 256, 0, stream>>>(av_fc_w, WfcvT, kEMB, kH);
  k_convT<<<cdiv((long)kEMB * kH, 256), 256, 0, stream>>>(au_fc_w, WfcuT, kEMB, kH);
  k_convT<<<cdiv((long)kH * kD, 256), 256, 0, stream>>>(av_a_w, WavT, kH, kD);
  k_convT<<<cdiv((long)kH * kD, 256), 256, 0, stream>>>(av_b_w, WbvT, kH, kD);
  k_convT<<<cdiv((long)kH * kD, 256), 256, 0, stream>>>(au_a_w, WauT, kH, kD);
  k_convT<<<cdiv((long)kH * kD, 256), 256, 0, stream>>>(au_b_w, WbuT, kH, kD);

  // 2) fused CLAM pipelines (WMMA + TDM double-buffered input staging)
  k_clam_v<<<cdiv(kN1, 64), 1024, SMEM_V, stream>>>(
      X, WvT, pv_b, WfcvT, av_fc_b, WavT, av_a_b, WbvT, av_b_b, av_c_w, av_c_b,
      hpv, Av, kN1);
  k_clam_u<<<cdiv(kN2, 64), 1024, SMEM_U, stream>>>(
      Y, WfcuT, au_fc_b, WauT, au_a_b, WbuT, au_b_b, au_c_w, au_c_b,
      hpu, Au, kN2);

  // 3) softmax-weighted pooling M = softmax(A) @ hp (deterministic reduction)
  const int rpc1 = cdiv(kN1, kChunks), rpc2 = cdiv(kN2, kChunks);
  k_colmax<<<kNC, 256, 0, stream>>>(Av, kN1, mxv);
  k_colmax<<<kNC, 256, 0, stream>>>(Au, kN2, mxu);
  k_attn_accum<<<kChunks, 256, 0, stream>>>(Av, mxv, hpv, kN1, rpc1, partv);
  k_attn_accum<<<kChunks, 256, 0, stream>>>(Au, mxu, hpu, kN2, rpc2, partu);
  k_reduce_part<<<cdiv(kNC * kH + kNC, 256), 256, 0, stream>>>(partv, Maccv, Zv);
  k_reduce_part<<<cdiv(kNC * kH + kNC, 256), 256, 0, stream>>>(partu, Maccu, Zu);

  // 4) gate + MoE + fusion + classifier head
  k_head<<<1, 256, 0, stream>>>(Maccv, Zv, Maccu, Zu, clinical, clin_b, clin_w,
                                g1_b, g1_w, g2_b, g2_w, e1l1_b, e1l1_w, e1l2_b,
                                e1l2_w, e2l1_b, e2l1_w, e2l2_b, e2l2_w, e2o_b,
                                e2o_w, e3l1_b, e3l1_w, e3l2_b, e3l2_w, fu_b, fu_w,
                                cls_b, cls_w, out);
}
